// HeteroGNN_51032801411221
// MI455X (gfx1250) — compile-verified, matrix operand order checked by reading, not run on hardware
//
#include <hip/hip_runtime.h>

#define NN 2048   // entities
#define HH 256    // hidden
#define RR 22     // relations
#define EE 20000  // edges per relation
#define KT (RR*HH + HH)   // 5888: fused GEMM K (22 mean blocks + root x block)

typedef __bf16 bf16_t;
typedef __attribute__((ext_vector_type(8)))  bf16_t v8bf;
typedef __attribute__((ext_vector_type(16))) bf16_t v16bf;
typedef __attribute__((ext_vector_type(8)))  float  v8f;
typedef __attribute__((ext_vector_type(2)))  float  vf2;
typedef __attribute__((ext_vector_type(4)))  int    v4i;

#if defined(__gfx1250__) && __has_builtin(__builtin_amdgcn_global_load_async_to_lds_b128)
#define USE_ASYNC_LDS 1
#else
#define USE_ASYNC_LDS 0
#endif

__device__ __forceinline__ void async_copy16(const bf16_t* g, bf16_t* l) {
#if USE_ASYNC_LDS
    __builtin_amdgcn_global_load_async_to_lds_b128(
        (__attribute__((address_space(1))) v4i*)g,
        (__attribute__((address_space(3))) v4i*)l,
        /*imm offset*/0, /*cpol*/0);
#else
    *(v8bf*)l = *(const v8bf*)g;
#endif
}

__device__ __forceinline__ void wait_async_copies() {
#if USE_ASYNC_LDS
#if __has_builtin(__builtin_amdgcn_s_wait_asynccnt)
    __builtin_amdgcn_s_wait_asynccnt(0);
#else
    asm volatile("s_wait_asynccnt 0x0" ::: "memory");
#endif
#endif
}

// ---------------------------------------------------------------- utilities
__global__ __launch_bounds__(256) void zero_kernel(float* __restrict__ p, long n) {
    long t = (long)blockIdx.x * blockDim.x + threadIdx.x;
    long stride = (long)gridDim.x * blockDim.x;
    for (; t < n; t += stride) p[t] = 0.0f;
}

// ------------------------------------------------- scatter: per-dst degree
__global__ __launch_bounds__(256) void scatter_cnt_kernel(const int* __restrict__ ei,
                                                          float* __restrict__ cnt) {
    int t = blockIdx.x * 256 + threadIdx.x;
    if (t >= RR * EE) return;
    int r = t / EE, e = t % EE;
    int dst = ei[((size_t)r * 2 + 1) * EE + e];
    unsafeAtomicAdd(&cnt[(size_t)r * NN + dst], 1.0f);
}

// --------------------------------------------- scatter: segment-sum (L2 atomics)
__global__ __launch_bounds__(256) void scatter_sum_kernel(const int* __restrict__ ei,
                                                          const float* __restrict__ xin,
                                                          float* __restrict__ sums) {
    long t = (long)blockIdx.x * 256 + threadIdx.x;
    if (t >= (long)RR * EE * (HH / 4)) return;
    int hg = (int)(t & 63);
    long te = t >> 6;                 // r*EE + e
    int r = (int)(te / EE);
    int e = (int)(te % EE);
    int src = ei[((size_t)r * 2 + 0) * EE + e];
    int dst = ei[((size_t)r * 2 + 1) * EE + e];
    const float4 v = *(const float4*)&xin[(size_t)src * HH + hg * 4];
    float* b = &sums[((size_t)r * NN + dst) * HH + hg * 4];
    unsafeAtomicAdd(b + 0, v.x);
    unsafeAtomicAdd(b + 1, v.y);
    unsafeAtomicAdd(b + 2, v.z);
    unsafeAtomicAdd(b + 3, v.w);
}

// ------------- build A = [mean_0 | ... | mean_21 | x]  (bf16, [NN][KT] row-major)
__global__ __launch_bounds__(256) void pack_a_kernel(const float* __restrict__ sums,
                                                     const float* __restrict__ cnt,
                                                     const float* __restrict__ xin,
                                                     bf16_t* __restrict__ A) {
    long t = (long)blockIdx.x * 256 + threadIdx.x;
    if (t >= (long)NN * KT) return;
    int n = (int)(t / KT);
    int k = (int)(t % KT);
    float val;
    if (k < RR * HH) {
        int r = k >> 8, h = k & 255;
        float c = cnt[(size_t)r * NN + n];
        float cc = c > 1.0f ? c : 1.0f;
        val = sums[((size_t)r * NN + n) * HH + h] / cc;
    } else {
        val = xin[(size_t)n * HH + (k - RR * HH)];
    }
    A[t] = (bf16_t)val;
}

// ------------- build B (bf16, [256 out-cols][KT]): B[n][k]
//   relation block r: Wl[l][r][n][k%HH];  root block: sum_r Wr[l][r][n][.]
__global__ __launch_bounds__(256) void pack_b_kernel(const float* __restrict__ Wl,
                                                     const float* __restrict__ Wr,
                                                     int l, bf16_t* __restrict__ B) {
    long t = (long)blockIdx.x * 256 + threadIdx.x;
    if (t >= (long)HH * KT) return;
    int n = (int)(t / KT);
    int k = (int)(t % KT);
    float val;
    if (k < RR * HH) {
        int r = k >> 8, kk = k & 255;
        val = Wl[(((size_t)l * RR + r) * HH + n) * HH + kk];
    } else {
        int kk = k - RR * HH;
        float acc = 0.0f;
        for (int r = 0; r < RR; ++r)
            acc += Wr[(((size_t)l * RR + r) * HH + n) * HH + kk];
        val = acc;
    }
    B[t] = (bf16_t)val;
}

__global__ __launch_bounds__(256) void pack_bias_kernel(const float* __restrict__ bl,
                                                        int l, float* __restrict__ bias) {
    int h = threadIdx.x;
    float acc = 0.0f;
    for (int r = 0; r < RR; ++r)
        acc += bl[((size_t)l * RR + r) * HH + h];
    bias[h] = acc;
}

// ---------------------------------------------------------------- WMMA GEMM
// out[M=2048,256] = relu( A[M,KT] @ B^T + bias ),  A bf16 row-major,
// B bf16 stored [n][KT].  Block tile 64x128, 8 waves (2x4), wave tile 32x32.
// Double-buffered LDS staged via GLOBAL_LOAD_ASYNC_TO_LDS (ASYNCcnt).
#define BM 64
#define BN 128
#define BK 32
__global__ __launch_bounds__(256) void gemm_bias_relu_kernel(const bf16_t* __restrict__ A,
                                                             const bf16_t* __restrict__ B,
                                                             const float* __restrict__ bias,
                                                             float* __restrict__ out,
                                                             int M, int K) {
    __shared__ bf16_t sA[2][BM * BK];
    __shared__ bf16_t sB[2][BN * BK];
    const int tid  = threadIdx.x;
    const int lane = tid & 31;
    const int wave = tid >> 5;
    const int waveM = wave & 1;   // 0..1 -> 32-row chunk
    const int waveN = wave >> 1;  // 0..3 -> 32-col chunk
    const int blockM = blockIdx.x * BM;
    const int blockN = blockIdx.y * BN;

    v8f c[2][2] = {};

    const int ldRow = tid >> 2;          // 0..63
    const int ldCol = (tid & 3) * 8;     // 0,8,16,24

    auto stage = [&](int buf, int kk) {
        async_copy16(&A[(size_t)(blockM + ldRow) * K + kk + ldCol],
                     &sA[buf][ldRow * BK + ldCol]);
        async_copy16(&B[(size_t)(blockN + ldRow) * K + kk + ldCol],
                     &sB[buf][ldRow * BK + ldCol]);
        async_copy16(&B[(size_t)(blockN + ldRow + 64) * K + kk + ldCol],
                     &sB[buf][(ldRow + 64) * BK + ldCol]);
    };

    stage(0, 0);
    int buf = 0;
    for (int kk = 0; kk < K; kk += BK) {
        wait_async_copies();
        __syncthreads();                 // current buffer visible to all waves
        if (kk + BK < K) stage(buf ^ 1, kk + BK);   // prefetch next tile

        // ISA bf16 A 16x32 layout: lane<16 -> K in [0,8)+[16,24); lane>=16 -> +8
        const int fr   = lane & 15;
        const int koff = (lane >> 4) * 8;
        #pragma unroll
        for (int mt = 0; mt < 2; ++mt) {
            const bf16_t* pa = &sA[buf][(waveM * 32 + mt * 16 + fr) * BK + koff];
            v8bf a0 = *(const v8bf*)pa;
            v8bf a1 = *(const v8bf*)(pa + 16);
            v16bf av = __builtin_shufflevector(a0, a1, 0,1,2,3,4,5,6,7,8,9,10,11,12,13,14,15);
            #pragma unroll
            for (int nt = 0; nt < 2; ++nt) {
                const bf16_t* pb = &sB[buf][(waveN * 32 + nt * 16 + fr) * BK + koff];
                v8bf b0 = *(const v8bf*)pb;
                v8bf b1 = *(const v8bf*)(pb + 16);
                v16bf bv = __builtin_shufflevector(b0, b1, 0,1,2,3,4,5,6,7,8,9,10,11,12,13,14,15);
                c[mt][nt] = __builtin_amdgcn_wmma_f32_16x16x32_bf16(
                    false, av, false, bv, (short)0, c[mt][nt], false, false);
            }
        }
        buf ^= 1;
    }

    // epilogue: C 16x16 f32 layout -> lanes 0-15: N=lane, M=j; lanes 16-31: N=lane-16, M=j+8
    const int nlane = lane & 15;
    const int mhalf = (lane >> 4) * 8;
    #pragma unroll
    for (int mt = 0; mt < 2; ++mt) {
        #pragma unroll
        for (int nt = 0; nt < 2; ++nt) {
            #pragma unroll
            for (int j = 0; j < 8; ++j) {
                int rowg = blockM + waveM * 32 + mt * 16 + mhalf + j;
                int colg = blockN + waveN * 32 + nt * 16 + nlane;
                float v = c[mt][nt][j] + bias[colg];
                out[(size_t)rowg * HH + colg] = v > 0.0f ? v : 0.0f;
            }
        }
    }
}

// ------------------------------- m = relu(x@Wm.T+bm), a = relu(x@Wa.T+ba), s = sum m*a
__global__ __launch_bounds__(64) void coeff_kernel(const float* __restrict__ xf,
                                                   const float* __restrict__ Wa,
                                                   const float* __restrict__ ba,
                                                   const float* __restrict__ Wm,
                                                   const float* __restrict__ bm,
                                                   float* __restrict__ m_out,
                                                   float* __restrict__ s_out) {
    __shared__ float xrow[HH];
    __shared__ float mv[RR], av[RR];
    const int n = blockIdx.x;
    for (int h = threadIdx.x; h < HH; h += 64) xrow[h] = xf[(size_t)n * HH + h];
    __syncthreads();
    const int t = threadIdx.x;
    if (t < RR) {
        float acc = bm[t];
        for (int h = 0; h < HH; ++h) acc += xrow[h] * Wm[(size_t)t * HH + h];
        mv[t] = acc > 0.0f ? acc : 0.0f;
    } else if (t >= 32 && t < 32 + RR) {
        int r = t - 32;
        float acc = ba[r];
        for (int h = 0; h < HH; ++h) acc += xrow[h] * Wa[(size_t)r * HH + h];
        av[r] = acc > 0.0f ? acc : 0.0f;
    }
    __syncthreads();
    if (t == 0) {
        float s = 0.0f;
        for (int r = 0; r < RR; ++r) s += mv[r] * av[r];
        s_out[n] = s;
    }
    if (t < RR) m_out[(size_t)n * RR + t] = mv[t];
}

// --------------- out[i,j] = s[i] + sum_r m[i,r]*adj[i,j,r]
// Streams 369MB of adj exactly once -> non-temporal loads/stores (no L2 reuse).
__global__ __launch_bounds__(256) void final_kernel(const float* __restrict__ adj,
                                                    const float* __restrict__ m,
                                                    const float* __restrict__ s,
                                                    float* __restrict__ out) {
    const int i = blockIdx.x;
    __shared__ float sm[RR + 2];
    if (threadIdx.x < RR) sm[threadIdx.x] = m[(size_t)i * RR + threadIdx.x];
    if (threadIdx.x == 31) sm[RR] = s[i];
    __syncthreads();
    float mloc[RR];
    #pragma unroll
    for (int r = 0; r < RR; ++r) mloc[r] = sm[r];
    const float sv = sm[RR];
    const vf2* arow = (const vf2*)(adj + (size_t)i * NN * RR);  // 88B per j, 8B aligned
    for (int j = threadIdx.x; j < NN; j += 256) {
        const vf2* p = arow + (size_t)j * (RR / 2);
        float acc = sv;
        #pragma unroll
        for (int r = 0; r < RR / 2; ++r) {
            vf2 v = __builtin_nontemporal_load(&p[r]);
            acc += mloc[2 * r] * v.x + mloc[2 * r + 1] * v.y;
        }
        __builtin_nontemporal_store(acc, &out[(size_t)i * NN + j]);
    }
}

// ---------------------------------------------------------------- launcher
extern "C" void kernel_launch(void* const* d_in, const int* in_sizes, int n_in,
                              void* d_out, int out_size, void* d_ws, size_t ws_size,
                              hipStream_t stream) {
    const float* x   = (const float*)d_in[0];
    const float* adj = (const float*)d_in[1];
    const float* Wl  = (const float*)d_in[2];
    const float* bl  = (const float*)d_in[3];
    const float* Wr  = (const float*)d_in[4];
    const float* Wa  = (const float*)d_in[5];
    const float* ba  = (const float*)d_in[6];
    const float* Wm  = (const float*)d_in[7];
    const float* bm  = (const float*)d_in[8];
    const int*   ei  = (const int*)d_in[9];
    (void)in_sizes; (void)n_in; (void)out_size; (void)ws_size;

    char* ws = (char*)d_ws;
    size_t off = 0;
    auto take = [&](size_t bytes) -> char* {
        char* p = ws + off;
        off += (bytes + 255) & ~(size_t)255;
        return p;
    };
    float*  sums  = (float*) take((size_t)RR * NN * HH * 4);
    float*  cnt   = (float*) take((size_t)RR * NN * 4);
    bf16_t* Apack = (bf16_t*)take((size_t)NN * KT * 2);
    bf16_t* Bpack = (bf16_t*)take((size_t)HH * KT * 2);
    float*  biasS = (float*) take((size_t)HH * 4);
    float*  x0    = (float*) take((size_t)NN * HH * 4);
    float*  x1    = (float*) take((size_t)NN * HH * 4);
    float*  mbuf  = (float*) take((size_t)NN * RR * 4);
    float*  sbuf  = (float*) take((size_t)NN * 4);

    const float* xin = x;
    for (int l = 0; l < 2; ++l) {
        float* xout = (l == 0) ? x0 : x1;
        zero_kernel<<<4096, 256, 0, stream>>>(sums, (long)RR * NN * HH);
        zero_kernel<<<176, 256, 0, stream>>>(cnt, (long)RR * NN);
        scatter_cnt_kernel<<<(RR * EE + 255) / 256, 256, 0, stream>>>(ei, cnt);
        {
            long total = (long)RR * EE * (HH / 4);
            scatter_sum_kernel<<<(int)((total + 255) / 256), 256, 0, stream>>>(ei, xin, sums);
        }
        {
            long total = (long)NN * KT;
            pack_a_kernel<<<(int)((total + 255) / 256), 256, 0, stream>>>(sums, cnt, xin, Apack);
        }
        {
            long total = (long)HH * KT;
            pack_b_kernel<<<(int)((total + 255) / 256), 256, 0, stream>>>(Wl, Wr, l, Bpack);
        }
        pack_bias_kernel<<<1, 256, 0, stream>>>(bl, l, biasS);
        gemm_bias_relu_kernel<<<dim3(NN / BM, HH / BN), 256, 0, stream>>>(
            Apack, Bpack, biasS, xout, NN, KT);
        xin = xout;
    }

    coeff_kernel<<<NN, 64, 0, stream>>>(x1, Wa, ba, Wm, bm, mbuf, sbuf);
    final_kernel<<<NN, 256, 0, stream>>>(adj, mbuf, sbuf, (float*)d_out);
}
